// MultiheadAttention_68496138436736
// MI455X (gfx1250) — compile-verified, matrix operand order checked by reading, not run on hardware
//
#include <hip/hip_runtime.h>
#include <hip/hip_bf16.h>
#include <stdint.h>

typedef __bf16 bf16;
typedef __attribute__((ext_vector_type(16))) __bf16 v16bf;
typedef __attribute__((ext_vector_type(8)))  __bf16 v8bf;
typedef __attribute__((ext_vector_type(8)))  float  v8f;
typedef unsigned int v4u __attribute__((ext_vector_type(4)));
typedef int          v4i __attribute__((ext_vector_type(4)));
typedef int          v8i __attribute__((ext_vector_type(8)));

static __device__ __forceinline__ bf16 f2bf(float f) {
    unsigned u = __builtin_bit_cast(unsigned, f);
    unsigned r = (u + 0x7fffu + ((u >> 16) & 1u)) >> 16;
    unsigned short s = (unsigned short)r;
    return __builtin_bit_cast(bf16, s);
}

static __device__ __forceinline__ v16bf combine16(v8bf lo, v8bf hi) {
    v16bf r;
#pragma unroll
    for (int i = 0; i < 8; ++i) { r[i] = lo[i]; r[i + 8] = hi[i]; }
    return r;
}

static __device__ __forceinline__ uint32_t lds_off(const void* p) {
    return (uint32_t)(uintptr_t)p;   // low 32 bits of generic LDS addr = offset
}

// gfx1250 async memory->LDS copy (16B), tracked by ASYNCcnt.
static __device__ __forceinline__ void async_copy_b128(uint32_t loff, const void* gaddr) {
    asm volatile("global_load_async_to_lds_b128 %0, %1, off"
                 :: "v"(loff), "v"(gaddr) : "memory");
}

// gfx1250 Tensor Data Mover: DMA a 2D tile (tile1 rows x tile0 elems, bf16)
// from a row-major tensor (row length tdim0, tdim1 rows, row stride stride0
// elements) starting at gaddr into LDS at byte offset loff (rows packed at
// tile0*2 bytes). D# packing per CDNA5 ISA section 8. This toolchain exposes
// the 6-argument builtin: (g0 x4, g1 x8, g2 x4, g3 x4, x8, cpol).
static __device__ __forceinline__ void tdm_load_2d(
    uint32_t loff, const void* gaddr,
    uint32_t tdim0, uint32_t tdim1, uint32_t stride0,
    uint32_t tile0, uint32_t tile1)
{
    const uint64_t ga = (uint64_t)(uintptr_t)gaddr;
    v4u g0;
    g0[0] = 1u;                                           // count=1 (valid user D#)
    g0[1] = loff;                                         // lds_addr
    g0[2] = (uint32_t)ga;                                 // global_addr[31:0]
    g0[3] = (uint32_t)((ga >> 32) & 0x01FFFFFFu) | (2u << 30);  // addr[56:32] | type=2
    v8i g1;
    g1[0] = (int)(1u << 16);                              // data_size=1 (2 bytes)
    g1[1] = (int)((tdim0 & 0xFFFFu) << 16);               // tensor_dim0[15:0]
    g1[2] = (int)(((tdim0 >> 16) & 0xFFFFu) | ((tdim1 & 0xFFFFu) << 16));
    g1[3] = (int)(((tdim1 >> 16) & 0xFFFFu) | ((tile0 & 0xFFFFu) << 16));
    g1[4] = (int)(tile1 & 0xFFFFu);                       // tile_dim1 (tile_dim2=0)
    g1[5] = (int)stride0;                                 // tensor_dim0_stride[31:0]
    g1[6] = 0;
    g1[7] = 0;
    v4i z4 = {0, 0, 0, 0};                                // groups 2/3: dims unused (2D)
    v8i z8 = {0, 0, 0, 0, 0, 0, 0, 0};
    __builtin_amdgcn_tensor_load_to_lds(g0, g1, z4, z4, z8, 0);
}

// ---------------------------------------------------------------------------
// fp32 -> bf16 elementwise conversion
// ---------------------------------------------------------------------------
__global__ void cvt_f32_bf16(const float* __restrict__ src, bf16* __restrict__ dst, int n) {
    int i = blockIdx.x * blockDim.x + threadIdx.x;
    int stride = gridDim.x * blockDim.x;
    for (; i < n; i += stride) dst[i] = f2bf(src[i]);
}

// fp32 [R][C] -> bf16 [C][R] (transpose during convert; weights only, tiny)
__global__ void cvt_transpose_f32_bf16(const float* __restrict__ src, bf16* __restrict__ dst,
                                       int R, int Cc) {
    int i = blockIdx.x * blockDim.x + threadIdx.x;
    int stride = gridDim.x * blockDim.x;
    const int total = R * Cc;
    for (; i < total; i += stride) {
        int r = i / Cc, c = i - r * Cc;
        dst[(size_t)c * R + r] = f2bf(src[i]);
    }
}

// ---------------------------------------------------------------------------
// Tiled bf16 WMMA GEMM: C[M,N] = A[M,K] @ BT[N,K]^T + bias
// B-tile staging: double-buffered global_load_async_to_lds_b128 row copies.
// BM=128 (8 waves x 16 rows), BN=64, BK=32. 256 threads.
// mode 0: write fp32 C row-major.
// mode 1: scatter epilogue -> q/k bf16 [H=12][S=4096][64], v bf16 [H][64][S=4096]
// ---------------------------------------------------------------------------
__global__ __launch_bounds__(256) void gemm_bf16_wmma(
    const bf16* __restrict__ A, const bf16* __restrict__ BT,
    const float* __restrict__ bias, float* __restrict__ C,
    bf16* __restrict__ qb, bf16* __restrict__ kb, bf16* __restrict__ vb,
    int M, int N, int K, int mode)
{
    __shared__ bf16 Bt[2][64 * 32];              // [n_local][k_local]
    const int tid  = threadIdx.x;
    const int w    = tid >> 5;
    const int lane = tid & 31;
    const int idx  = lane & 15;
    const int half = lane >> 4;
    const int m0 = blockIdx.y * 128;
    const int n0 = blockIdx.x * 64;

    v8f acc[4];
#pragma unroll
    for (int j = 0; j < 4; ++j)
#pragma unroll
        for (int v = 0; v < 8; ++v) acc[j][v] = 0.0f;

    const bf16* Arow = A + (size_t)(m0 + w * 16 + idx) * K;

    // Staging: thread t copies 16B chunk (t&3) of row (t>>2)
    const int nrow = tid >> 2, nchk = tid & 3;
    const bf16* Bsrc = BT + (size_t)(n0 + nrow) * K + nchk * 8;
    const uint32_t bl0 = lds_off(&Bt[0][nrow * 32 + nchk * 8]);
    const uint32_t bl1 = lds_off(&Bt[1][nrow * 32 + nchk * 8]);

    auto compute = [&](int kb0, int p) {
        const bf16* ab = Arow + kb0;
        v16bf afr = combine16(*reinterpret_cast<const v8bf*>(ab + half * 8),
                              *reinterpret_cast<const v8bf*>(ab + 16 + half * 8));
#pragma unroll
        for (int j = 0; j < 4; ++j) {
            const int n = j * 16 + idx;
            v16bf bfr = combine16(
                *reinterpret_cast<const v8bf*>(&Bt[p][n * 32 + half * 8]),
                *reinterpret_cast<const v8bf*>(&Bt[p][n * 32 + 16 + half * 8]));
            acc[j] = __builtin_amdgcn_wmma_f32_16x16x32_bf16(
                false, afr, false, bfr, (short)0, acc[j], false, false);
        }
    };

    async_copy_b128(bl0, Bsrc);                  // prefetch tile 0
    int p = 0;
    int kb0 = 0;
    for (; kb0 + 32 < K; kb0 += 32) {            // steady state: next always issued
        async_copy_b128(p ? bl0 : bl1, Bsrc + kb0 + 32);
        asm volatile("s_wait_asynccnt 0x1" ::: "memory");
        __syncthreads();
        compute(kb0, p);
        __syncthreads();
        p ^= 1;
    }
    asm volatile("s_wait_asynccnt 0x0" ::: "memory");   // peeled last tile
    __syncthreads();
    compute(kb0, p);

    // Epilogue
#pragma unroll
    for (int j = 0; j < 4; ++j) {
#pragma unroll
        for (int v = 0; v < 8; ++v) {
            const int m = m0 + w * 16 + v + 8 * half;
            const int n = n0 + j * 16 + idx;
            float val = acc[j][v] + bias[n];
            if (mode == 0) {
                C[(size_t)m * N + n] = val;
            } else {
                const int which = n / 768;       // 0=q 1=k 2=v
                const int hc = n % 768;
                const int h = hc >> 6, d = hc & 63;
                if (which == 2) {                // V stored head-transposed [H][64][S]
                    vb[((size_t)(h * 64 + d)) * 4096 + m] = f2bf(val);
                } else {
                    bf16* dst = (which == 0) ? qb : kb;
                    dst[((size_t)(h * 4096 + m)) * 64 + d] = f2bf(val);
                }
            }
        }
    }
}

// ---------------------------------------------------------------------------
// Causal flash attention, one block per (64 q-rows, head). 128 threads = 4 waves,
// each wave owns 16 q rows. K/V tiles DMA'd by the Tensor Data Mover (wave 0
// issues two D# per tile, double-buffered, TENSORcnt-tracked).
// No 1/sqrt(hd) scaling (matches reference).
// ---------------------------------------------------------------------------
__global__ __launch_bounds__(128) void flash_attn_wmma(
    const bf16* __restrict__ qb, const bf16* __restrict__ kb,
    const bf16* __restrict__ vb, bf16* __restrict__ outb)
{
    constexpr int Sq = 4096, HD = 64, Dm = 768;
    __shared__ bf16 Kt[2][64 * 64];              // [key_local][d]
    __shared__ bf16 Vt[2][64 * 64];              // [d][key_local] (vb pre-transposed)
    __shared__ bf16 Pld[4 * 16 * 64];            // per-wave P scratch

    const int tid  = threadIdx.x;
    const int w    = tid >> 5;
    const int lane = tid & 31;
    const int idx  = lane & 15;
    const int half = lane >> 4;
    const int qt = blockIdx.x;                   // q tile (64 rows)
    const int h  = blockIdx.y;
    const int q0w = qt * 64 + w * 16;
    const size_t headOff = (size_t)h * Sq * HD;

    // TDM staging: K tile = 64 rows x 64 elems of [Sq x 64]; V tile = 64 rows x
    // 64 elems of the head-transposed [64 x Sq] matrix.
    auto stage = [&](int j, int buf) {
        if (w == 0) {
            tdm_load_2d(lds_off(&Kt[buf][0]),
                        kb + headOff + (size_t)j * 64 * HD,
                        HD, Sq, HD, 64, 64);
            tdm_load_2d(lds_off(&Vt[buf][0]),
                        vb + (size_t)(h * 64) * Sq + j * 64,
                        Sq, HD, Sq, 64, 64);
        }
    };

    // Load this wave's Q fragments once (16 rows x 64 dims -> two 16x32 A-frags)
    v16bf qa[2];
    {
        const bf16* qrow = qb + headOff + (size_t)(q0w + idx) * HD;
#pragma unroll
        for (int t = 0; t < 2; ++t)
            qa[t] = combine16(
                *reinterpret_cast<const v8bf*>(qrow + t * 32 + half * 8),
                *reinterpret_cast<const v8bf*>(qrow + t * 32 + 16 + half * 8));
    }

    v8f oacc[4];
    float mrow[8], lrow[8];
#pragma unroll
    for (int j = 0; j < 4; ++j)
#pragma unroll
        for (int v = 0; v < 8; ++v) oacc[j][v] = 0.0f;
#pragma unroll
    for (int v = 0; v < 8; ++v) { mrow[v] = -3.0e38f; lrow[v] = 0.0f; }

    auto process_tile = [&](int j, int p, bool domask) {
        // Scores S = Q @ K^T : 16 x 64 per wave
        v8f sacc[4];
#pragma unroll
        for (int j2 = 0; j2 < 4; ++j2)
#pragma unroll
            for (int v = 0; v < 8; ++v) sacc[j2][v] = 0.0f;
#pragma unroll
        for (int t = 0; t < 2; ++t) {
#pragma unroll
            for (int j2 = 0; j2 < 4; ++j2) {
                const int key = j2 * 16 + idx;
                v16bf bfr = combine16(
                    *reinterpret_cast<const v8bf*>(&Kt[p][key * 64 + t * 32 + half * 8]),
                    *reinterpret_cast<const v8bf*>(&Kt[p][key * 64 + t * 32 + 16 + half * 8]));
                sacc[j2] = __builtin_amdgcn_wmma_f32_16x16x32_bf16(
                    false, qa[t], false, bfr, (short)0, sacc[j2], false, false);
            }
        }

        if (domask) {                            // causal mask (diagonal tile only)
#pragma unroll
            for (int j2 = 0; j2 < 4; ++j2)
#pragma unroll
                for (int v = 0; v < 8; ++v) {
                    const int kg = j * 64 + j2 * 16 + idx;
                    const int qg = q0w + v + 8 * half;
                    sacc[j2][v] = (kg > qg) ? -1.0e30f : sacc[j2][v];
                }
        }

        // Online softmax per row (a row lives across the 16 lanes of one half)
#pragma unroll
        for (int v = 0; v < 8; ++v) {
            float rm = sacc[0][v];
#pragma unroll
            for (int j2 = 1; j2 < 4; ++j2) rm = fmaxf(rm, sacc[j2][v]);
#pragma unroll
            for (int off = 1; off < 16; off <<= 1)
                rm = fmaxf(rm, __shfl_xor(rm, off, 32));
            const float mnew  = fmaxf(mrow[v], rm);
            const float scale = __expf(mrow[v] - mnew);
            float ps = 0.0f;
#pragma unroll
            for (int j2 = 0; j2 < 4; ++j2) {
                const float pe = __expf(sacc[j2][v] - mnew);
                sacc[j2][v] = pe;
                ps += pe;
            }
#pragma unroll
            for (int off = 1; off < 16; off <<= 1)
                ps += __shfl_xor(ps, off, 32);
            lrow[v] = lrow[v] * scale + ps;
            mrow[v] = mnew;
#pragma unroll
            for (int j2 = 0; j2 < 4; ++j2) oacc[j2][v] *= scale;
        }

        // P (C-layout) -> LDS -> A-layout bf16 fragments (per-wave scratch;
        // same-wave LDS ops are in order, no barrier needed)
        {
            bf16* P = &Pld[w * 16 * 64];
#pragma unroll
            for (int j2 = 0; j2 < 4; ++j2)
#pragma unroll
                for (int v = 0; v < 8; ++v)
                    P[(v + 8 * half) * 64 + j2 * 16 + idx] = f2bf(sacc[j2][v]);
        }

        // O += P @ V
#pragma unroll
        for (int t2 = 0; t2 < 2; ++t2) {
            const bf16* prow = &Pld[w * 16 * 64 + idx * 64 + t2 * 32];
            v16bf pa = combine16(
                *reinterpret_cast<const v8bf*>(prow + half * 8),
                *reinterpret_cast<const v8bf*>(prow + 16 + half * 8));
#pragma unroll
            for (int j2 = 0; j2 < 4; ++j2) {
                const int d = j2 * 16 + idx;
                v16bf bfr = combine16(
                    *reinterpret_cast<const v8bf*>(&Vt[p][d * 64 + t2 * 32 + half * 8]),
                    *reinterpret_cast<const v8bf*>(&Vt[p][d * 64 + t2 * 32 + 16 + half * 8]));
                oacc[j2] = __builtin_amdgcn_wmma_f32_16x16x32_bf16(
                    false, pa, false, bfr, (short)0, oacc[j2], false, false);
            }
        }
    };

    stage(0, 0);
    int p = 0;
    for (int j = 0; j < qt; ++j) {               // non-diagonal tiles, no mask
        stage(j + 1, p ^ 1);                     // 2 TDM descriptors in flight on top
        if (w == 0) __builtin_amdgcn_s_wait_tensorcnt(2);   // current tile done
        __syncthreads();
        process_tile(j, p, false);
        __syncthreads();                         // readers done before next DMA lands
        p ^= 1;
    }
    if (w == 0) __builtin_amdgcn_s_wait_tensorcnt(0);       // peeled diagonal tile
    __syncthreads();
    process_tile(qt, p, true);

    // Normalize and write bf16 output [S, D] (head-interleaved columns)
#pragma unroll
    for (int j2 = 0; j2 < 4; ++j2)
#pragma unroll
        for (int v = 0; v < 8; ++v) {
            const int s = q0w + v + 8 * half;
            const int d = j2 * 16 + idx;
            outb[(size_t)s * Dm + h * 64 + d] = f2bf(oacc[j2][v] / lrow[v]);
        }
}

// ---------------------------------------------------------------------------
// Host launcher
// ---------------------------------------------------------------------------
extern "C" void kernel_launch(void* const* d_in, const int* in_sizes, int n_in,
                              void* d_out, int out_size, void* d_ws, size_t ws_size,
                              hipStream_t stream) {
    (void)in_sizes; (void)n_in; (void)out_size; (void)ws_size;
    constexpr int S = 4096, D = 768, H = 12;

    const float* x      = (const float*)d_in[0];  // [S, D]
    const float* W_attn = (const float*)d_in[1];  // [D, 3D]
    const float* b_attn = (const float*)d_in[2];  // [3D]
    const float* W_proj = (const float*)d_in[3];  // [D, D]
    const float* b_proj = (const float*)d_in[4];  // [D]
    float* out = (float*)d_out;                   // [S, D]

    char* ws = (char*)d_ws;
    size_t off = 0;
    auto carve = [&](size_t bytes) { char* p = ws + off; off += (bytes + 255) & ~(size_t)255; return p; };
    bf16* xb   = (bf16*)carve((size_t)S * D * 2);          // x bf16 [S][D]
    bf16* WabT = (bf16*)carve((size_t)D * 3 * D * 2);      // W_attn^T bf16 [3D][D]
    bf16* WpbT = (bf16*)carve((size_t)D * D * 2);          // W_proj^T bf16 [D][D]
    bf16* qb   = (bf16*)carve((size_t)H * S * 64 * 2);     // Q [H][S][64]
    bf16* kbuf = (bf16*)carve((size_t)H * S * 64 * 2);     // K [H][S][64]
    bf16* vbuf = (bf16*)carve((size_t)H * S * 64 * 2);     // V [H][64][S] (transposed)
    bf16* atb  = (bf16*)carve((size_t)S * D * 2);          // attention out bf16 [S][D]

    // 1) fp32 -> bf16 conversions (weights transposed so GEMM B-staging is a
    //    straight async row copy)
    cvt_f32_bf16<<<1024, 256, 0, stream>>>(x, xb, S * D);
    cvt_transpose_f32_bf16<<<1024, 256, 0, stream>>>(W_attn, WabT, D, 3 * D);
    cvt_transpose_f32_bf16<<<512,  256, 0, stream>>>(W_proj, WpbT, D, D);

    // 2) QKV projection GEMM: [S,D] @ [D,3D] + b -> q/k head-major, v head-transposed
    {
        dim3 grid(3 * D / 64, S / 128);
        gemm_bf16_wmma<<<grid, 256, 0, stream>>>(
            xb, WabT, b_attn, nullptr, qb, kbuf, vbuf, S, 3 * D, D, /*mode=*/1);
    }

    // 3) Causal flash attention per (q-tile, head), TDM-staged K/V
    {
        dim3 grid(S / 64, H);
        flash_attn_wmma<<<grid, 128, 0, stream>>>(qb, kbuf, vbuf, atb);
    }

    // 4) Output projection GEMM: [S,D] @ [D,D] + b -> fp32 out
    {
        dim3 grid(D / 64, S / 128);
        gemm_bf16_wmma<<<grid, 256, 0, stream>>>(
            atb, WpbT, b_proj, out, nullptr, nullptr, nullptr, S, D, D, /*mode=*/0);
    }
}
